// MRT12_Layer_26594437497334
// MI455X (gfx1250) — matrix operand order repaired and checked by reading, not compiled, not run on hardware
//
#include <hip/hip_runtime.h>
#include <hip/hip_bf16.h>
#include <stdint.h>

#define D_MODEL 1024
#define E_DIM   2048
#define SEQ_T   4096
#define BATCH   4
#define N_TOK   (BATCH*SEQ_T)      /* 16384 */
#define CHUNK   128
#define NCHUNK  (SEQ_T/CHUNK)      /* 32 */
#define PI_F    3.14159265358979323846f
#define BSTRIDE 72                 /* LDS row stride in bf16 elems (64 + 8 pad) */
#define SLICE_ELEMS (128 * BSTRIDE)            /* one staged W slice (bf16 elems) */
#define SLICE_BYTES (SLICE_ELEMS * 2)

typedef __attribute__((ext_vector_type(16))) __bf16       v16bf;
typedef __attribute__((ext_vector_type(8)))  float        v8f;
typedef __attribute__((ext_vector_type(4)))  unsigned int v4u;

union Frag {
  v16bf bf;
  v4u   q[2];
};

__device__ __forceinline__ unsigned short f2bf(float f) {
  unsigned int u = __float_as_uint(f);
  unsigned int r = u + 0x7FFFu + ((u >> 16) & 1u);   // round-to-nearest-even
  return (unsigned short)(r >> 16);
}

__device__ __forceinline__ v8f wmma_bf16(v16bf a, v16bf b, v8f c) {
  // (neg_a, A, neg_b, B, c_mod, C, reuse_a, reuse_b)
  return __builtin_amdgcn_wmma_f32_16x16x32_bf16(false, a, false, b, (short)0, c,
                                                 false, false);
}

// Issue one per-lane 16B async copy global -> LDS (ASYNCcnt-tracked).
__device__ __forceinline__ void async_g2l_b128(unsigned int lds_addr,
                                               const unsigned short* gptr) {
  asm volatile("global_load_async_to_lds_b128 %0, %1, off"
               :: "v"(lds_addr), "v"((unsigned long long)(uintptr_t)gptr)
               : "memory");
}
__device__ __forceinline__ void wait_asynccnt0() {
  asm volatile("s_wait_asynccnt 0" ::: "memory");
}

// ---------------------------------------------------------------------------
// fp32 -> bf16 conversion
// ---------------------------------------------------------------------------
__global__ __launch_bounds__(256)
void cvt_bf16_kernel(const float* __restrict__ in, unsigned short* __restrict__ o, int n)
{
  int i = blockIdx.x * 256 + threadIdx.x;
  if (i < n) o[i] = f2bf(in[i]);
}

// ---------------------------------------------------------------------------
// bf16 WMMA GEMM: OUT[t,o] = epilogue( sum_k A[t,k] * W[o,k] )
// A: [M=16384, K] bf16 row-major, W: [N, K] bf16 row-major.
// Block tile 128x128 (8 waves of 32x64).
// W k-slices (128 x 64) double-buffered in LDS via GLOBAL_LOAD_ASYNC_TO_LDS_B128:
// slice k+1 is issued before computing slice k, so the async copy is hidden
// under the 16 WMMAs + A-loads of the current slice.
// MODE 0: store fp32 raw (GEMM1 -> res)
// MODE 1: +bias, SiLU, store bf16 (GEMM2 -> hid)
// MODE 2: +bias, store fp32 (GEMM3 -> h_final)
// ---------------------------------------------------------------------------
template<int MODE>
__global__ __launch_bounds__(256)
void gemm_bf16_kernel(const unsigned short* __restrict__ A,
                      const unsigned short* __restrict__ W,
                      void* __restrict__ OUT,
                      const float* __restrict__ bias,
                      int N, int K)
{
  __shared__ unsigned short sB[2 * SLICE_ELEMS];   // 36 KB double-buffered slice

  const int lane = threadIdx.x & 31;
  const int wave = threadIdx.x >> 5;
  const int wr   = wave >> 1;        // 0..3  (t-direction)
  const int wc   = wave & 1;         // 0..1  (o-direction)
  const int ll   = lane & 15;
  const int hi   = lane >> 4;

  const int tBase  = blockIdx.y * 128 + wr * 32;
  const int blockO = blockIdx.x * 128;
  const int oBase  = blockO + wc * 64;

  v8f acc[2][4];
  #pragma unroll
  for (int i = 0; i < 2; ++i)
    #pragma unroll
    for (int j = 0; j < 4; ++j)
      acc[i][j] = (v8f){0.f,0.f,0.f,0.f,0.f,0.f,0.f,0.f};

  const unsigned short* aR[2];
  aR[0] = A + (size_t)(tBase + ll)      * K;
  aR[1] = A + (size_t)(tBase + 16 + ll) * K;

  // Per-thread staging assignment: 4 x 16B chunks of the 128x64 W slice.
  const unsigned short* gW[4];
  unsigned int lAddr[4];
  #pragma unroll
  for (int i = 0; i < 4; ++i) {
    const int cid = threadIdx.x + 256 * i;       // 0..1023
    const int row = cid >> 3;                    // 0..127
    const int ce  = (cid & 7) * 8;               // element offset in slice row
    gW[i]    = W + (size_t)(blockO + row) * K + ce;
    lAddr[i] = (unsigned int)(uintptr_t)(&sB[row * BSTRIDE + ce]);
  }

  const int KSTEPS = K >> 6;                     // 64-wide k slices

  // Prologue: stage slice 0 into buffer 0.
  #pragma unroll
  for (int i = 0; i < 4; ++i) async_g2l_b128(lAddr[i], gW[i]);
  wait_asynccnt0();
  __syncthreads();

  for (int s = 0; s < KSTEPS; ++s) {
    const int k0 = s << 6;
    const unsigned cur = (unsigned)(s & 1);

    // Kick off the next slice into the other buffer before computing.
    if (s + 1 < KSTEPS) {
      const unsigned nxt = (unsigned)((s + 1) & 1) * SLICE_BYTES;
      #pragma unroll
      for (int i = 0; i < 4; ++i)
        async_g2l_b128(lAddr[i] + nxt, gW[i] + k0 + 64);
      #pragma unroll
      for (int i = 0; i < 2; ++i) __builtin_prefetch(aR[i] + k0 + 64, 0, 3);
    }

    const unsigned sbase = cur * SLICE_ELEMS;    // element offset of live buffer

    #pragma unroll
    for (int ks = 0; ks < 2; ++ks) {
      const int ksub = ks * 32;
      Frag a[2], b[4];
      const int ka = k0 + ksub + 8 * hi;  // A: K {0..7,16..23}(lo)/{8..15,24..31}(hi)
      #pragma unroll
      for (int i = 0; i < 2; ++i) {
        a[i].q[0] = *(const v4u*)(aR[i] + ka);
        a[i].q[1] = *(const v4u*)(aR[i] + ka + 16);
      }
      #pragma unroll
      for (int j = 0; j < 4; ++j) {       // B: K {0..7,16..23}(lo)/{8..15,24..31}(hi)
        const unsigned short* p =
            &sB[sbase + (wc * 64 + 16 * j + ll) * BSTRIDE + ksub + 8 * hi];
        b[j].q[0] = *(const v4u*)(p);
        b[j].q[1] = *(const v4u*)(p + 16);
      }
      #pragma unroll
      for (int i = 0; i < 2; ++i)
        #pragma unroll
        for (int j = 0; j < 4; ++j)
          acc[i][j] = wmma_bf16(a[i].bf, b[j].bf, acc[i][j]);
    }

    // Retire this wave's in-flight copy, then make all copies visible.
    if (s + 1 < KSTEPS) {
      wait_asynccnt0();
      __syncthreads();
    }
  }

  // C/D layout: VGPR r, lane n -> row (r + 8*(n>=16)), col (n&15)
  #pragma unroll
  for (int i = 0; i < 2; ++i) {
    #pragma unroll
    for (int j = 0; j < 4; ++j) {
      #pragma unroll
      for (int r = 0; r < 8; ++r) {
        const int row = tBase + 16*i + r + 8*hi;
        const int col = oBase + 16*j + ll;
        float v = acc[i][j][r];
        const size_t o = (size_t)row * N + col;
        if (MODE == 0) {
          ((float*)OUT)[o] = v;
        } else if (MODE == 1) {
          v += bias[col];
          const float s2 = v * (1.0f / (1.0f + __expf(-v)));
          ((unsigned short*)OUT)[o] = f2bf(s2);
        } else {
          ((float*)OUT)[o] = v + bias[col];
        }
      }
    }
  }
}

// ---------------------------------------------------------------------------
// Scan phase 1: per (batch, chunk) block, activate coefficients in place
//   b_t = sigmoid(alpha)*tanh(theta)*pi   (stored into theta slot)
//   a_t = 1 - sigmoid(alpha)              (stored into alpha slot)
// and accumulate the chunk-composed affine (A,B): h_out = A*h_in + B.
// ---------------------------------------------------------------------------
__global__ __launch_bounds__(256)
void scan_compose_kernel(float* __restrict__ res,
                         float* __restrict__ compA,
                         float* __restrict__ compB)
{
  const int chunk = blockIdx.x;
  const int b     = blockIdx.y;
  const int t0    = chunk * CHUNK;
  float Aacc[4] = {1.f,1.f,1.f,1.f};
  float Bacc[4] = {0.f,0.f,0.f,0.f};
  for (int t = 0; t < CHUNK; ++t) {
    const size_t base = (size_t)(b*SEQ_T + t0 + t) * E_DIM;
    #pragma unroll
    for (int j = 0; j < 4; ++j) {
      const int c = threadIdx.x + 256*j;
      const float th  = res[base + c];
      const float al  = res[base + D_MODEL + c];
      const float sig = 1.0f / (1.0f + __expf(-al));
      const float bt  = sig * tanhf(th) * PI_F;
      const float at  = 1.0f - sig;
      res[base + c]           = bt;
      res[base + D_MODEL + c] = at;
      Aacc[j] *= at;
      Bacc[j]  = fmaf(at, Bacc[j], bt);
    }
  }
  const size_t cb = (size_t)(b*NCHUNK + chunk) * D_MODEL;
  #pragma unroll
  for (int j = 0; j < 4; ++j) {
    const int c = threadIdx.x + 256*j;
    compA[cb + c] = Aacc[j];
    compB[cb + c] = Bacc[j];
  }
}

// ---------------------------------------------------------------------------
// Scan phase 2: serial carry across the 32 chunks, one thread per (b, channel)
// ---------------------------------------------------------------------------
__global__ __launch_bounds__(256)
void scan_carry_kernel(const float* __restrict__ compA,
                       const float* __restrict__ compB,
                       float* __restrict__ carry)
{
  const int idx = blockIdx.x * 256 + threadIdx.x;   // 0..4095
  const int b   = idx >> 10;
  const int c   = idx & (D_MODEL - 1);
  float h = 0.f;
  for (int k = 0; k < NCHUNK; ++k) {
    const size_t o = (size_t)(b*NCHUNK + k) * D_MODEL + c;
    carry[o] = h;
    h = fmaf(compA[o], h, compB[o]);
  }
}

// ---------------------------------------------------------------------------
// Scan phase 3: re-apply recurrence with chunk-entry carry, emit h (fp32)
// ---------------------------------------------------------------------------
__global__ __launch_bounds__(256)
void scan_apply_kernel(const float* __restrict__ res,
                       const float* __restrict__ carry,
                       float* __restrict__ hbuf)
{
  const int chunk = blockIdx.x;
  const int b     = blockIdx.y;
  const int t0    = chunk * CHUNK;
  const size_t cb = (size_t)(b*NCHUNK + chunk) * D_MODEL;
  float h[4];
  #pragma unroll
  for (int j = 0; j < 4; ++j) h[j] = carry[cb + threadIdx.x + 256*j];
  for (int t = 0; t < CHUNK; ++t) {
    const size_t base  = (size_t)(b*SEQ_T + t0 + t) * E_DIM;
    const size_t obase = (size_t)(b*SEQ_T + t0 + t) * D_MODEL;
    #pragma unroll
    for (int j = 0; j < 4; ++j) {
      const int c  = threadIdx.x + 256*j;
      const float at = res[base + D_MODEL + c];
      const float bt = res[base + c];
      h[j] = fmaf(at, h[j], bt);
      hbuf[obase + c] = h[j];
    }
  }
}

// ---------------------------------------------------------------------------
// Depthwise causal conv (k=3, left pad 2) + residual; emit z in bf16
// ---------------------------------------------------------------------------
__global__ __launch_bounds__(256)
void conv_z_kernel(const float* __restrict__ hbuf,
                   const float* __restrict__ cw,
                   const float* __restrict__ cbias,
                   unsigned short* __restrict__ zb)
{
  const long long idx = (long long)blockIdx.x * 256 + threadIdx.x;
  const int c  = (int)(idx & (D_MODEL - 1));
  const long long tg = idx >> 10;
  const int tb = (int)(tg & (SEQ_T - 1));           // position within sequence
  const float h0  = hbuf[idx];
  const float hm1 = (tb >= 1) ? hbuf[idx - D_MODEL]     : 0.f;
  const float hm2 = (tb >= 2) ? hbuf[idx - 2*D_MODEL]   : 0.f;
  float conv = cbias[c];
  conv = fmaf(hm2, cw[3*c + 0], conv);
  conv = fmaf(hm1, cw[3*c + 1], conv);
  conv = fmaf(h0,  cw[3*c + 2], conv);
  zb[idx] = f2bf(h0 + conv);
}

// ---------------------------------------------------------------------------
// RMS norm over d=1024, one block per token
// ---------------------------------------------------------------------------
__global__ __launch_bounds__(256)
void rms_norm_kernel(const float* __restrict__ hf,
                     const float* __restrict__ nw,
                     float* __restrict__ out)
{
  __shared__ float red[256];
  const size_t tok = blockIdx.x;
  const float* xrow = hf + tok * D_MODEL;
  float v[4];
  float s = 0.f;
  #pragma unroll
  for (int j = 0; j < 4; ++j) {
    v[j] = xrow[threadIdx.x + 256*j];
    s = fmaf(v[j], v[j], s);
  }
  red[threadIdx.x] = s;
  __syncthreads();
  for (int st = 128; st > 0; st >>= 1) {
    if (threadIdx.x < st) red[threadIdx.x] += red[threadIdx.x + st];
    __syncthreads();
  }
  const float scale = rsqrtf(red[0] * (1.0f / (float)D_MODEL) + 1e-6f);
  #pragma unroll
  for (int j = 0; j < 4; ++j) {
    const int c = threadIdx.x + 256*j;
    out[tok*D_MODEL + c] = v[j] * scale * nw[c];
  }
}

// ---------------------------------------------------------------------------
// Host-side launch. Workspace layout (lifetime-overlapped):
//   [0,128M)    res (fp32)  -> later: zb bf16 [0,32M), hid bf16 [32M,96M)
//   [128M,192M) hbuf (fp32) -> later: h_final (fp32)
//   [192M,224M) xb  bf16
//   [224M,228M) wkb bf16, [228M,232M) w1b, [232M,236M) w2b
//   [236M, +1.5M) compA/compB/carry
// ---------------------------------------------------------------------------
extern "C" void kernel_launch(void* const* d_in, const int* in_sizes, int n_in,
                              void* d_out, int out_size, void* d_ws, size_t ws_size,
                              hipStream_t stream)
{
  (void)in_sizes; (void)n_in; (void)out_size; (void)ws_size;

  const float* x      = (const float*)d_in[0];
  const float* w_key  = (const float*)d_in[1];
  const float* conv_w = (const float*)d_in[2];
  const float* conv_b = (const float*)d_in[3];
  const float* w1     = (const float*)d_in[4];
  const float* b1     = (const float*)d_in[5];
  const float* w2     = (const float*)d_in[6];
  const float* b2     = (const float*)d_in[7];
  const float* norm_w = (const float*)d_in[8];
  float* out = (float*)d_out;

  char* ws = (char*)d_ws;
  float*          res   = (float*)(ws);
  unsigned short* zb    = (unsigned short*)(ws);
  unsigned short* hid   = (unsigned short*)(ws + (32ull  << 20));
  float*          hbuf  = (float*)(ws + (128ull << 20));
  float*          hfin  = (float*)(ws + (128ull << 20));
  unsigned short* xb    = (unsigned short*)(ws + (192ull << 20));
  unsigned short* wkb   = (unsigned short*)(ws + (224ull << 20));
  unsigned short* w1b   = (unsigned short*)(ws + (228ull << 20));
  unsigned short* w2b   = (unsigned short*)(ws + (232ull << 20));
  float*          compA = (float*)(ws + (236ull << 20));
  float*          compB = (float*)(ws + (236ull << 20) + (1ull << 19));
  float*          carry = (float*)(ws + (236ull << 20) + (2ull << 19));

  const dim3 blk(256);

  // fp32 -> bf16 conversions
  cvt_bf16_kernel<<<(N_TOK*D_MODEL)/256, blk, 0, stream>>>(x,     xb,  N_TOK*D_MODEL);
  cvt_bf16_kernel<<<(E_DIM*D_MODEL)/256, blk, 0, stream>>>(w_key, wkb, E_DIM*D_MODEL);
  cvt_bf16_kernel<<<(E_DIM*D_MODEL)/256, blk, 0, stream>>>(w1,    w1b, E_DIM*D_MODEL);
  cvt_bf16_kernel<<<(D_MODEL*E_DIM)/256, blk, 0, stream>>>(w2,    w2b, D_MODEL*E_DIM);

  // GEMM1: res = x @ w_key^T   (16384 x 2048, K=1024)
  gemm_bf16_kernel<0><<<dim3(E_DIM/128, N_TOK/128), blk, 0, stream>>>(
      xb, wkb, (void*)res, nullptr, E_DIM, D_MODEL);

  // chunked linear-recurrence scan
  scan_compose_kernel<<<dim3(NCHUNK, BATCH), blk, 0, stream>>>(res, compA, compB);
  scan_carry_kernel<<<dim3((BATCH*D_MODEL)/256), blk, 0, stream>>>(compA, compB, carry);
  scan_apply_kernel<<<dim3(NCHUNK, BATCH), blk, 0, stream>>>(res, carry, hbuf);

  // depthwise causal conv + residual -> z (bf16)
  conv_z_kernel<<<(N_TOK*D_MODEL)/256, blk, 0, stream>>>(hbuf, conv_w, conv_b, zb);

  // GEMM2: hid = silu(z @ w1^T + b1)   (16384 x 2048, K=1024) -> bf16
  gemm_bf16_kernel<1><<<dim3(E_DIM/128, N_TOK/128), blk, 0, stream>>>(
      zb, w1b, (void*)hid, b1, E_DIM, D_MODEL);

  // GEMM3: h_final = hid @ w2^T + b2   (16384 x 1024, K=2048) -> fp32
  gemm_bf16_kernel<2><<<dim3(D_MODEL/128, N_TOK/128), blk, 0, stream>>>(
      hid, w2b, (void*)hfin, b2, D_MODEL, E_DIM);

  // RMS norm
  rms_norm_kernel<<<dim3(N_TOK), blk, 0, stream>>>(hfin, norm_w, out);
}
